// SwinTransformerBlock_33672543600799
// MI455X (gfx1250) — matrix operand order repaired and checked by reading, not compiled, not run on hardware
//
#include <hip/hip_runtime.h>
#include <hip/hip_bf16.h>

typedef __attribute__((ext_vector_type(16))) _Float16 v16h;
typedef __attribute__((ext_vector_type(8)))  _Float16 v8h;
typedef __attribute__((ext_vector_type(4)))  _Float16 v4h;
typedef __attribute__((ext_vector_type(8)))  float    v8f;

// ---------------------------------------------------------------------------
// Problem constants (B=2, D=32, H=64, W=64, C=128, window 4x4x4, shift 2)
// ---------------------------------------------------------------------------
#define CCH   128
#define NTOK  64            // tokens per window
#define TOT_TOK (262144)    // 2*32*64*64

// ---------------------------------------------------------------------------
// WMMA fragment helpers (CDNA5 wave32 layouts, f16 16x16x32)
// A (16x32, row-major src, leading dim ld in halves):
//   lane<16 : row M=lane, elems 0..7 -> K=0..7,  elems 8..15 -> K=16..23
//   lane>=16: row M=lane-16, elems 0..7 -> K=8..15, elems 8..15 -> K=24..31
// B (32x16, row-major src): N = lane&15, K order same as A (g*8+e, 16+g*8+e-8)
// C/D (16x16 f32): VGPR r -> M = r + (lane>=16 ? 8 : 0), N = lane & 15
// ---------------------------------------------------------------------------
__device__ __forceinline__ v16h load_frag_a(const _Float16* p, int ld) {
    int lane = threadIdx.x & 31;
    int m = lane & 15;
    int g = (lane >> 4) & 1;
    const _Float16* row = p + m * ld + g * 8;
    v8h lo = *(const v8h*)(row);        // K = g*8 .. g*8+7
    v8h hi = *(const v8h*)(row + 16);   // K = 16+g*8 .. 16+g*8+7
    return __builtin_shufflevector(lo, hi, 0,1,2,3,4,5,6,7,8,9,10,11,12,13,14,15);
}

__device__ __forceinline__ v16h load_frag_b(const _Float16* p, int ld) {
    int lane = threadIdx.x & 31;
    int n = lane & 15;
    int g = (lane >> 4) & 1;
    v16h b;
    #pragma unroll
    for (int s = 0; s < 8; ++s) {
        b[s]     = p[(g * 8 + s) * ld + n];
        b[s + 8] = p[(16 + g * 8 + s) * ld + n];
    }
    return b;
}

__device__ __forceinline__ v8f wmma_f16(v16h a, v16h b, v8f c) {
    return __builtin_amdgcn_wmma_f32_16x16x32_f16(false, a, false, b,
                                                  (short)0, c, false, false);
}

__device__ __forceinline__ float gelu_tanh(float x) {
    const float k0 = 0.7978845608028654f;   // sqrt(2/pi)
    const float k1 = 0.044715f;
    float t = tanhf(k0 * (x + k1 * x * x * x));
    return 0.5f * x * (1.0f + t);
}

// meshgrid('xy') relative-position index, replicating the reference exactly.
__device__ __forceinline__ int relidx(int p, int q) {
    int ip = (p >> 2) & 3, jp = p >> 4, kp = p & 3;
    int iq = (q >> 2) & 3, jq = q >> 4, kq = q & 3;
    return ((ip - iq + 3) * 7 + (jp - jq + 3)) * 7 + (kp - kq + 3);
}

// ---------------------------------------------------------------------------
// Kernel 0a: fp32 -> f16 weight conversion
// ---------------------------------------------------------------------------
__global__ void k_cvt(const float* __restrict__ a, _Float16* __restrict__ o, int n) {
    int i = blockIdx.x * 256 + threadIdx.x;
    if (i < n) o[i] = (_Float16)a[i];
}

// ---------------------------------------------------------------------------
// Kernel 0b: expand rel-pos bias table -> biasx[h][p][q] (4 x 64 x 64 f32).
// Window-invariant, so computed exactly once (64 KB, L2-resident afterwards).
// ---------------------------------------------------------------------------
__global__ void k_bias(const float* __restrict__ relt, float* __restrict__ biasx) {
    int i = blockIdx.x * 256 + threadIdx.x;   // 0 .. 16383
    int h = i >> 12;
    int p = (i >> 6) & 63;
    int q = i & 63;
    biasx[i] = relt[relidx(p, q) * 4 + h];
}

// ---------------------------------------------------------------------------
// Kernel 1: LN1 + cyclic shift(-2) + window partition -> xw (bw*64, 128) f16
// One wave per token: 32 lanes x float4 covers C=128.
// ---------------------------------------------------------------------------
__global__ __launch_bounds__(256)
void k_ln1_partition(const float* __restrict__ x,
                     const float* __restrict__ g1, const float* __restrict__ b1,
                     _Float16* __restrict__ xw) {
    int wave = threadIdx.x >> 5;
    int lane = threadIdx.x & 31;
    long gt = (long)blockIdx.x * 8 + wave;      // global window-token slot
    int g    = (int)(gt >> 6);
    int t    = (int)(gt & 63);
    int bb   = g >> 11;
    int wlin = g & 2047;
    int zd = wlin >> 8, yh = (wlin >> 4) & 15, xq = wlin & 15;
    int td = t >> 4, th = (t >> 2) & 3, tw = t & 3;
    int oz = (zd * 4 + td + 2) & 31;            // roll(-2): read at +2
    int oy = (yh * 4 + th + 2) & 63;
    int ox = (xq * 4 + tw + 2) & 63;
    const float* src = x + ((((long)bb * 32 + oz) * 64 + oy) * 64 + ox) * CCH;

    float4 v = *(const float4*)(src + lane * 4);
    float s  = v.x + v.y + v.z + v.w;
    float ss = v.x * v.x + v.y * v.y + v.z * v.z + v.w * v.w;
    #pragma unroll
    for (int off = 16; off > 0; off >>= 1) {
        s  += __shfl_xor(s, off);
        ss += __shfl_xor(ss, off);
    }
    float mean = s * (1.0f / 128.0f);
    float inv  = rsqrtf(ss * (1.0f / 128.0f) - mean * mean + 1e-6f);
    float4 gg = *(const float4*)(g1 + lane * 4);
    float4 bv = *(const float4*)(b1 + lane * 4);
    v4h o;
    o[0] = (_Float16)((v.x - mean) * inv * gg.x + bv.x);
    o[1] = (_Float16)((v.y - mean) * inv * gg.y + bv.y);
    o[2] = (_Float16)((v.z - mean) * inv * gg.z + bv.z);
    o[3] = (_Float16)((v.w - mean) * inv * gg.w + bv.w);
    *(v4h*)(xw + gt * CCH + lane * 4) = o;
}

// ---------------------------------------------------------------------------
// Kernel 2: per-window attention, 8 waves / block, 60 KB LDS.
//   qkv (per head, via f16 WMMA) -> QK^T (+bias+mask) -> softmax -> AV
//   -> proj -> reverse shift scatter + shortcut residual -> d_out (x1)
// ---------------------------------------------------------------------------
__global__ __launch_bounds__(256)
void k_attn(const _Float16* __restrict__ xw,
            const _Float16* __restrict__ wqkv,    // 128x384 f16
            const _Float16* __restrict__ wproj,   // 128x128 f16
            const float* __restrict__ mask,       // 2048x64x64
            const float* __restrict__ biasx,      // 4x64x64 (expanded)
            const float* __restrict__ projb,      // 128
            const float* __restrict__ x0,         // shortcut
            float* __restrict__ out) {            // x1
    __shared__ __align__(32) _Float16 xin[NTOK * CCH];   // 16 KB
    __shared__ __align__(32) _Float16 qh[NTOK * 32];     //  4 KB (scaled q)
    __shared__ __align__(32) _Float16 khT[32 * NTOK];    //  4 KB (k transposed)
    __shared__ __align__(32) _Float16 vh[NTOK * 32];     //  4 KB
    __shared__ __align__(32) float    S[NTOK * NTOK];    // 16 KB (f32 then f16)
    __shared__ __align__(32) _Float16 out16[NTOK * CCH]; // 16 KB

    int tid  = threadIdx.x;
    int wave = tid >> 5, lane = tid & 31;
    int g = blockIdx.x;
    int bb = g >> 11, wlin = g & 2047;

    // stage windowed activations into LDS; prefetch next window's block
    {
        const uint4* src = (const uint4*)(xw + (long)g * NTOK * CCH);
        uint4* dst = (uint4*)xin;
        if (g + 1 < 4096)   // pull next window toward WGP while we compute
            __builtin_prefetch(xw + (long)(g + 1) * NTOK * CCH + tid * 32, 0, 3);
        #pragma unroll
        for (int i = tid; i < NTOK * CCH / 8; i += 256) dst[i] = src[i];
    }
    __syncthreads();

    const float scale = 0.17677669529663687f;  // 1/sqrt(32)

    for (int h = 0; h < 4; ++h) {
        // ---- Q/K/V GEMMs: 8 tiles each, one tile per wave per matrix ----
        {
            int mt = wave >> 1, nt = wave & 1;
            #pragma unroll
            for (int jmat = 0; jmat < 3; ++jmat) {
                int col0 = jmat * 128 + h * 32 + nt * 16;
                v8f acc = {};
                #pragma unroll
                for (int kk = 0; kk < 4; ++kk) {
                    v16h a  = load_frag_a(xin + mt * 16 * CCH + kk * 32, CCH);
                    v16h bf = load_frag_b(wqkv + (kk * 32) * 384 + col0, 384);
                    acc = wmma_f16(a, bf, acc);
                }
                int m0 = mt * 16 + ((lane >> 4) << 3);
                int n  = lane & 15;
                if (jmat == 0) {
                    #pragma unroll
                    for (int r = 0; r < 8; ++r)
                        qh[(m0 + r) * 32 + nt * 16 + n] = (_Float16)(acc[r] * scale);
                } else if (jmat == 1) {
                    #pragma unroll
                    for (int r = 0; r < 8; ++r)     // store K transposed (d,q)
                        khT[(nt * 16 + n) * NTOK + m0 + r] = (_Float16)acc[r];
                } else {
                    #pragma unroll
                    for (int r = 0; r < 8; ++r)
                        vh[(m0 + r) * 32 + nt * 16 + n] = (_Float16)acc[r];
                }
            }
        }
        __syncthreads();

        // ---- S = q @ k^T  (+ rel-pos bias + shift mask fused in C-store) ----
        {
            const float* mrow = mask + (long)wlin * 4096;
            const float* brow = biasx + h * 4096;
            for (int tIdx = wave; tIdx < 16; tIdx += 8) {
                int mt = tIdx >> 2, nt = tIdx & 3;
                v16h a  = load_frag_a(qh + mt * 16 * 32, 32);        // K = 32
                v16h bf = load_frag_b(khT + nt * 16, NTOK);          // lane&15 = key q, K = d
                v8f acc = {};
                acc = wmma_f16(a, bf, acc);
                int m0 = mt * 16 + ((lane >> 4) << 3);
                int n  = nt * 16 + (lane & 15);
                #pragma unroll
                for (int r = 0; r < 8; ++r) {
                    int p = m0 + r;
                    S[p * NTOK + n] = acc[r] + brow[p * NTOK + n] + mrow[p * NTOK + n];
                }
            }
        }
        __syncthreads();

        // ---- row softmax (wave handles 8 rows, 2 cols/lane) ----
        #pragma unroll
        for (int r = 0; r < 8; ++r) {
            int row = wave * 8 + r;
            float a0 = S[row * NTOK + lane];
            float a1 = S[row * NTOK + 32 + lane];
            float mx = fmaxf(a0, a1);
            #pragma unroll
            for (int off = 16; off > 0; off >>= 1) mx = fmaxf(mx, __shfl_xor(mx, off));
            float e0 = __expf(a0 - mx), e1 = __expf(a1 - mx);
            float sum = e0 + e1;
            #pragma unroll
            for (int off = 16; off > 0; off >>= 1) sum += __shfl_xor(sum, off);
            float inv = 1.0f / sum;
            S[row * NTOK + lane]      = e0 * inv;
            S[row * NTOK + 32 + lane] = e1 * inv;
        }
        __syncthreads();

        // ---- in-place f32 -> f16 (register staging across a barrier) ----
        {
            float tmp[16];
            #pragma unroll
            for (int i = 0; i < 16; ++i) tmp[i] = S[tid * 16 + i];
            __syncthreads();
            _Float16* S16 = (_Float16*)S;
            #pragma unroll
            for (int i = 0; i < 16; ++i) S16[tid * 16 + i] = (_Float16)tmp[i];
        }
        __syncthreads();

        // ---- out_h = softmax(S) @ V : 8 tiles, one per wave ----
        {
            const _Float16* S16 = (const _Float16*)S;
            int mt = wave >> 1, nt = wave & 1;
            v8f acc = {};
            #pragma unroll
            for (int kk = 0; kk < 2; ++kk) {
                v16h a  = load_frag_a(S16 + mt * 16 * NTOK + kk * 32, NTOK);
                v16h bf = load_frag_b(vh + kk * 32 * 32 + nt * 16, 32);
                acc = wmma_f16(a, bf, acc);
            }
            int m0 = mt * 16 + ((lane >> 4) << 3);
            int n  = nt * 16 + (lane & 15);
            #pragma unroll
            for (int r = 0; r < 8; ++r)
                out16[(m0 + r) * CCH + h * 32 + n] = (_Float16)acc[r];
        }
        __syncthreads();
    }

    // ---- proj + bias + reverse-shift scatter + shortcut residual ----
    int zd = wlin >> 8, yh = (wlin >> 4) & 15, xq = wlin & 15;
    for (int tIdx = wave; tIdx < 32; tIdx += 8) {
        int mt = tIdx >> 3, nt = tIdx & 7;
        v8f acc = {};
        #pragma unroll
        for (int kk = 0; kk < 4; ++kk) {
            v16h a  = load_frag_a(out16 + mt * 16 * CCH + kk * 32, CCH);
            v16h bf = load_frag_b(wproj + (kk * 32) * CCH + nt * 16, CCH);
            acc = wmma_f16(a, bf, acc);
        }
        int m0 = mt * 16 + ((lane >> 4) << 3);
        int n  = nt * 16 + (lane & 15);
        float pb = projb[n];
        #pragma unroll
        for (int r = 0; r < 8; ++r) {
            int t  = m0 + r;
            int td = t >> 4, th = (t >> 2) & 3, tw = t & 3;
            int oz = (zd * 4 + td + 2) & 31;
            int oy = (yh * 4 + th + 2) & 63;
            int ox = (xq * 4 + tw + 2) & 63;
            long idx = ((((long)bb * 32 + oz) * 64 + oy) * 64 + ox) * CCH + n;
            out[idx] = x0[idx] + acc[r] + pb;
        }
    }
}

// ---------------------------------------------------------------------------
// Kernel 3: fused LN2 + MLP (fc1 + GELU + fc2) + residual, in-place on d_out.
// One block per 64-token tile; fc2 accumulators live in registers (4 v8f/wave).
// ---------------------------------------------------------------------------
__global__ __launch_bounds__(256)
void k_mlp(float* __restrict__ xio,
           const float* __restrict__ g2, const float* __restrict__ b2n,
           const _Float16* __restrict__ w1, const float* __restrict__ bias1,
           const _Float16* __restrict__ w2, const float* __restrict__ bias2) {
    __shared__ __align__(32) _Float16 xn[64 * CCH];   // 16 KB (LN2 output, f16)
    __shared__ __align__(32) _Float16 hc[64 * 128];   // 16 KB (hidden chunk, f16)

    int tid = threadIdx.x, wave = tid >> 5, lane = tid & 31;
    long row0 = (long)blockIdx.x * 64;

    // ---- LN2 over the 64 rows ----
    #pragma unroll
    for (int r = 0; r < 8; ++r) {
        int row = wave * 8 + r;
        const float* src = xio + (row0 + row) * CCH;
        float4 v = *(const float4*)(src + lane * 4);
        float s  = v.x + v.y + v.z + v.w;
        float ss = v.x * v.x + v.y * v.y + v.z * v.z + v.w * v.w;
        #pragma unroll
        for (int off = 16; off > 0; off >>= 1) {
            s  += __shfl_xor(s, off);
            ss += __shfl_xor(ss, off);
        }
        float mean = s * (1.0f / 128.0f);
        float inv  = rsqrtf(ss * (1.0f / 128.0f) - mean * mean + 1e-6f);
        float4 gg = *(const float4*)(g2 + lane * 4);
        float4 bv = *(const float4*)(b2n + lane * 4);
        v4h o;
        o[0] = (_Float16)((v.x - mean) * inv * gg.x + bv.x);
        o[1] = (_Float16)((v.y - mean) * inv * gg.y + bv.y);
        o[2] = (_Float16)((v.z - mean) * inv * gg.z + bv.z);
        o[3] = (_Float16)((v.w - mean) * inv * gg.w + bv.w);
        *(v4h*)(xn + row * CCH + lane * 4) = o;
    }
    __syncthreads();

    // per-wave fc2 output accumulators: tiles wave*4 .. wave*4+3 of (4M x 8N)
    v8f acc[4];
    #pragma unroll
    for (int j = 0; j < 4; ++j) { v8f z = {}; acc[j] = z; }

    for (int c = 0; c < 4; ++c) {                 // hidden chunks of 128
        // fc1 + GELU -> hc
        for (int tIdx = wave; tIdx < 32; tIdx += 8) {
            int mt = tIdx >> 3, nt = tIdx & 7;
            v8f a2 = {};
            #pragma unroll
            for (int kk = 0; kk < 4; ++kk) {
                v16h a  = load_frag_a(xn + mt * 16 * CCH + kk * 32, CCH);
                v16h bf = load_frag_b(w1 + (kk * 32) * 512 + c * 128 + nt * 16, 512);
                a2 = wmma_f16(a, bf, a2);
            }
            int m0 = mt * 16 + ((lane >> 4) << 3);
            int n  = nt * 16 + (lane & 15);
            float bb = bias1[c * 128 + n];
            #pragma unroll
            for (int r = 0; r < 8; ++r)
                hc[(m0 + r) * 128 + n] = (_Float16)gelu_tanh(a2[r] + bb);
        }
        __syncthreads();
        // fc2 partial accumulation (K = this chunk's 128 hidden)
        #pragma unroll
        for (int j = 0; j < 4; ++j) {
            int tIdx = wave * 4 + j;
            int mt = tIdx >> 3, nt = tIdx & 7;
            #pragma unroll
            for (int kk = 0; kk < 4; ++kk) {
                v16h a  = load_frag_a(hc + mt * 16 * 128 + kk * 32, 128);
                v16h bf = load_frag_b(w2 + ((long)c * 128 + kk * 32) * CCH + nt * 16, CCH);
                acc[j] = wmma_f16(a, bf, acc[j]);
            }
        }
        __syncthreads();
    }

    // ---- epilogue: + bias2 + residual (x1 read-modify-write) ----
    #pragma unroll
    for (int j = 0; j < 4; ++j) {
        int tIdx = wave * 4 + j;
        int mt = tIdx >> 3, nt = tIdx & 7;
        int m0 = mt * 16 + ((lane >> 4) << 3);
        int n  = nt * 16 + (lane & 15);
        float bb = bias2[n];
        #pragma unroll
        for (int r = 0; r < 8; ++r) {
            long idx = (row0 + m0 + r) * CCH + n;
            xio[idx] = xio[idx] + acc[j][r] + bb;
        }
    }
}

// ---------------------------------------------------------------------------
// Host launcher
// ---------------------------------------------------------------------------
extern "C" void kernel_launch(void* const* d_in, const int* in_sizes, int n_in,
                              void* d_out, int out_size, void* d_ws, size_t ws_size,
                              hipStream_t stream) {
    const float* x     = (const float*)d_in[0];
    const float* mask  = (const float*)d_in[1];
    const float* n1s   = (const float*)d_in[2];
    const float* n1b   = (const float*)d_in[3];
    const float* qkvw  = (const float*)d_in[4];
    const float* relt  = (const float*)d_in[5];
    const float* projw = (const float*)d_in[6];
    const float* projb = (const float*)d_in[7];
    const float* n2s   = (const float*)d_in[8];
    const float* n2b   = (const float*)d_in[9];
    const float* w1    = (const float*)d_in[10];
    const float* b1    = (const float*)d_in[11];
    const float* w2    = (const float*)d_in[12];
    const float* b2    = (const float*)d_in[13];
    float* out = (float*)d_out;

    char* ws = (char*)d_ws;
    _Float16* wqkv16  = (_Float16*)(ws);             // 49152 halves
    _Float16* wproj16 = (_Float16*)(ws + 98304);     // 16384 halves
    _Float16* w1_16   = (_Float16*)(ws + 131072);    // 65536 halves
    _Float16* w2_16   = (_Float16*)(ws + 262144);    // 65536 halves
    float*    biasx   = (float*)   (ws + 393216);    // 4*64*64 f32 (64 KB)
    _Float16* xw      = (_Float16*)(ws + 524288);    // 262144*128 halves (64 MB)

    k_cvt<<<192, 256, 0, stream>>>(qkvw,  wqkv16,  49152);
    k_cvt<<<64,  256, 0, stream>>>(projw, wproj16, 16384);
    k_cvt<<<256, 256, 0, stream>>>(w1,    w1_16,   65536);
    k_cvt<<<256, 256, 0, stream>>>(w2,    w2_16,   65536);
    k_bias<<<64, 256, 0, stream>>>(relt, biasx);

    k_ln1_partition<<<TOT_TOK / 8, 256, 0, stream>>>(x, n1s, n1b, xw);
    k_attn<<<4096, 256, 0, stream>>>(xw, wqkv16, wproj16, mask, biasx, projb, x, out);
    k_mlp<<<4096, 256, 0, stream>>>(out, n2s, n2b, w1_16, b1, w2_16, b2);
}
